// SparseMoeBlock_75514114998539
// MI455X (gfx1250) — matrix operand order repaired and checked by reading, not compiled, not run on hardware
//
#include <hip/hip_runtime.h>
#include <hip/hip_bf16.h>
#include <stdint.h>

#define HIDDEN 1024
#define INTER  512
#define NEXP   16
#define TTOT   4096   // 2 * 2048 tokens

typedef __attribute__((ext_vector_type(16))) __bf16        v16bf;
typedef __attribute__((ext_vector_type(8)))  float         v8f;
typedef __attribute__((ext_vector_type(8)))  unsigned int  v8u;

union Frag {
    v8u   u;
    v16bf b;
};

// ---- fp32 -> bf16 round-to-nearest-even (used only in one-time passes) ----
__device__ __forceinline__ unsigned int bf16_rne(float f) {
    unsigned int u = __float_as_uint(f);
    return (u + 0x7FFFu + ((u >> 16) & 1u)) >> 16;
}
__device__ __forceinline__ unsigned int bf16_pk_rne(float lo, float hi) {
    return bf16_rne(lo) | (bf16_rne(hi) << 16);
}
// ---- cheap truncation pack (fallback hot path only): 2 VALU per pair ----
__device__ __forceinline__ unsigned int bf16_pk_trunc(float lo, float hi) {
    return (__float_as_uint(lo) >> 16) | (__float_as_uint(hi) & 0xFFFF0000u);
}

// B fragment from fp32 weights with truncation packing (fallback path).
__device__ __forceinline__ Frag load_b_f32(const float* __restrict__ p) {
    Frag f;
    const float4* q = (const float4*)p;
#pragma unroll
    for (int i = 0; i < 4; ++i) {
        float4 v = q[i];
        f.u[2 * i]     = bf16_pk_trunc(v.x, v.y);
        f.u[2 * i + 1] = bf16_pk_trunc(v.z, v.w);
    }
    return f;
}

// A fragment from an LDS row of packed bf16 pairs.
// Lane L holds row M = L&15. Lanes 0-15: K = kb+{0..15},
// lanes 16-31: K = kb+{16..31}  (contiguous K, matches B fragments).
// `p` points at row_base + (kb + (lane>>4)*16)/2 in uint units.
__device__ __forceinline__ Frag load_a_lds(const unsigned int* p) {
    Frag f;
#pragma unroll
    for (int i = 0; i < 4; ++i) {
        f.u[i]     = p[i];
        f.u[4 + i] = p[4 + i];
    }
    return f;
}

__device__ __forceinline__ v8f wmma_bf16(Frag a, Frag b, v8f c) {
    return __builtin_amdgcn_wmma_f32_16x16x32_bf16(
        /*neg_a=*/false, a.b, /*neg_b=*/false, b.b,
        /*c_mod=*/(short)0, c, /*reuse_a=*/false, /*reuse_b=*/false);
}

// ---------------------------------------------------------------------------
// Kernel: one-time fp32 -> bf16 weight conversion (RNE). Memory bound.
// ---------------------------------------------------------------------------
__global__ void moe_cvt_kernel(const float4* __restrict__ src,
                               uint2* __restrict__ dst, int n4) {
    int i = blockIdx.x * blockDim.x + threadIdx.x;
    if (i < n4) {
        float4 v = src[i];
        dst[i] = make_uint2(bf16_pk_rne(v.x, v.y), bf16_pk_rne(v.z, v.w));
    }
}

// ---------------------------------------------------------------------------
// Kernel: zero output (we accumulate with f32 atomics) + expert counters.
// ---------------------------------------------------------------------------
__global__ void moe_zero_kernel(float* __restrict__ out, int* __restrict__ cnt) {
    size_t i = (size_t)blockIdx.x * blockDim.x + threadIdx.x;
    if (i < (size_t)TTOT * HIDDEN / 4) {
        ((float4*)out)[i] = make_float4(0.f, 0.f, 0.f, 0.f);
    }
    if (blockIdx.x == 0 && threadIdx.x < NEXP) cnt[threadIdx.x] = 0;
}

// ---------------------------------------------------------------------------
// Kernel: router. One wave32 per token: 16 logits, top-2, normalized top-2
// softmax weights, bucket (token, weight) per expert (makes experts sparse).
// ---------------------------------------------------------------------------
__global__ __launch_bounds__(256) void moe_router_kernel(
    const float* __restrict__ x, const float* __restrict__ rw,
    int* __restrict__ cnt, int* __restrict__ tok, float* __restrict__ wgt) {
    const int lane = threadIdx.x & 31;
    const int wv   = threadIdx.x >> 5;
    const int t    = blockIdx.x * 8 + wv;   // grid = TTOT/8 blocks

    float acc[NEXP];
#pragma unroll
    for (int e = 0; e < NEXP; ++e) acc[e] = 0.f;

    const float* xr = x + (size_t)t * HIDDEN;
    for (int d = lane; d < HIDDEN; d += 32) {
        float xv = xr[d];
#pragma unroll
        for (int e = 0; e < NEXP; ++e) acc[e] += xv * rw[e * HIDDEN + d];
    }
#pragma unroll
    for (int e = 0; e < NEXP; ++e) {
#pragma unroll
        for (int off = 16; off >= 1; off >>= 1)
            acc[e] += __shfl_xor(acc[e], off, 32);
    }

    if (lane == 0) {
        float m0 = -3.4e38f; int i0 = 0;
#pragma unroll
        for (int e = 0; e < NEXP; ++e)
            if (acc[e] > m0) { m0 = acc[e]; i0 = e; }
        float m1 = -3.4e38f; int i1 = 0;
#pragma unroll
        for (int e = 0; e < NEXP; ++e)
            if (e != i0 && acc[e] > m1) { m1 = acc[e]; i1 = e; }

        float s  = __expf(m1 - m0);        // <= 1, numerically safe
        float w0 = 1.0f / (1.0f + s);
        float w1 = s * w0;

        int p0 = atomicAdd(&cnt[i0], 1);
        tok[i0 * TTOT + p0] = t;  wgt[i0 * TTOT + p0] = w0;
        int p1 = atomicAdd(&cnt[i1], 1);
        tok[i1 * TTOT + p1] = t;  wgt[i1 * TTOT + p1] = w1;
    }
}

// ---------------------------------------------------------------------------
// Kernel: sparse expert MLP. One block = (expert e, tile of 16 routed tokens),
// 8 waves. Phase1: gate/up GEMM (bf16 WMMA, f32 acc) + silu -> LDS h tile.
// Phase2: down GEMM, scale by routing weight, atomic accumulate into out.
// BF16W=true: weights are preconverted bf16 (hot loop has ZERO conversion
// VALU; B loads are 2x global_load_b128 per fragment straight from L2).
// BF16W=false: fallback, inline truncation conversion from fp32.
// ---------------------------------------------------------------------------
template <bool BF16W>
__global__ __launch_bounds__(256) void moe_expert_kernel(
    const float* __restrict__ x, const void* __restrict__ gup,
    const void* __restrict__ down, const int* __restrict__ cnt,
    const int* __restrict__ tok, const float* __restrict__ wgt,
    float* __restrict__ out) {
    __shared__ unsigned int   xlds[16 * 512];   // 16 x 1024 bf16 pairs, 32 KB
    __shared__ unsigned short hlds[16 * 512];   // 16 x 512  bf16, 16 KB
    __shared__ int            tokbuf[16];
    __shared__ float          wbuf[16];

    const int e  = blockIdx.x >> 8;     // 256 worst-case tiles per expert
    const int tb = blockIdx.x & 255;
    const int nt = cnt[e];
    if (tb * 16 >= nt) return;          // uniform early-exit for idle tiles

    const int tid  = threadIdx.x;
    const int lane = tid & 31;
    const int wv   = tid >> 5;
    const int n    = lane & 15;         // B column / A row / D column
    const int kh   = lane >> 4;         // which K half this lane covers

    if (tid < 16) {
        int idx = tb * 16 + tid;
        if (idx < nt) { tokbuf[tid] = tok[e * TTOT + idx]; wbuf[tid] = wgt[e * TTOT + idx]; }
        else          { tokbuf[tid] = 0;                   wbuf[tid] = 0.0f; }
    }
    __syncthreads();

    // ---- stage gathered x rows into LDS as bf16 (coalesced float4 reads) ----
    {
        const int r = tid >> 4;         // row 0..15, 16 threads per row
        const int c = tid & 15;
        const float   wr  = wbuf[r];
        const float4* src = (const float4*)(x + (size_t)tokbuf[r] * HIDDEN);
#pragma unroll
        for (int i = 0; i < 16; ++i) {
            int c4 = c + 16 * i;        // 256 float4 per row
            float4 v = (wr != 0.0f) ? src[c4] : make_float4(0.f, 0.f, 0.f, 0.f);
            xlds[r * 512 + c4 * 2]     = bf16_pk_rne(v.x, v.y);
            xlds[r * 512 + c4 * 2 + 1] = bf16_pk_rne(v.z, v.w);
        }
    }
    __syncthreads();

    // ---- Phase 1: gu = x @ W^T, h = silu(g)*u (paired gate/up per wave) ----
    const unsigned short* g16 = (const unsigned short*)gup + ((size_t)e << 20);
    const float*          g32 = (const float*)gup + ((size_t)e << 20);
    for (int it = wv; it < INTER / 16; it += 8) {          // 32 column tiles of 16
        v8f accg = {0.f,0.f,0.f,0.f,0.f,0.f,0.f,0.f};
        v8f accu = {0.f,0.f,0.f,0.f,0.f,0.f,0.f,0.f};
        const size_t rowoff = (size_t)(it * 16 + n) * HIDDEN + kh * 16;
        if (it + 8 < INTER / 16) {
            const size_t nxt = (size_t)((it + 8) * 16 + n) * HIDDEN;
            if constexpr (BF16W) __builtin_prefetch(g16 + nxt, 0, 0);
            else                 __builtin_prefetch(g32 + nxt, 0, 0);
        }
#pragma unroll 4
        for (int kb = 0; kb < HIDDEN; kb += 32) {
            Frag a = load_a_lds(&xlds[n * 512 + ((kb + kh * 16) >> 1)]);
            Frag bg, bu;
            if constexpr (BF16W) {
                bg.u = *(const v8u*)(g16 + rowoff + kb);
                bu.u = *(const v8u*)(g16 + (size_t)INTER * HIDDEN + rowoff + kb);
            } else {
                bg = load_b_f32(g32 + rowoff + kb);
                bu = load_b_f32(g32 + (size_t)INTER * HIDDEN + rowoff + kb);
            }
            accg = wmma_bf16(a, bg, accg);
            accu = wmma_bf16(a, bu, accu);
        }
#pragma unroll
        for (int r = 0; r < 8; ++r) {
            int   m = r + 8 * kh;                          // token row (D layout)
            float g = accg[r];
            float h = (g / (1.0f + __expf(-g))) * accu[r]; // silu(g) * u
            hlds[m * 512 + it * 16 + n] = (unsigned short)bf16_rne(h);
        }
    }
    __syncthreads();

    // ---- Phase 2: o = h @ Wd^T, out[token] += w * o ------------------------
    const unsigned short* d16 = (const unsigned short*)down + ((size_t)e << 19);
    const float*          d32 = (const float*)down + ((size_t)e << 19);
    const unsigned int* hview = (const unsigned int*)hlds; // row stride 256 uints
    for (int dt = wv; dt < HIDDEN / 16; dt += 8) {         // 64 column tiles of 16
        v8f acc = {0.f,0.f,0.f,0.f,0.f,0.f,0.f,0.f};
        const size_t rowoff = (size_t)(dt * 16 + n) * INTER + kh * 16;
        if (dt + 8 < HIDDEN / 16) {
            const size_t nxt = (size_t)((dt + 8) * 16 + n) * INTER;
            if constexpr (BF16W) __builtin_prefetch(d16 + nxt, 0, 0);
            else                 __builtin_prefetch(d32 + nxt, 0, 0);
        }
#pragma unroll 4
        for (int kb = 0; kb < INTER; kb += 32) {
            Frag a = load_a_lds(hview + n * 256 + ((kb + kh * 16) >> 1));
            Frag b;
            if constexpr (BF16W) b.u = *(const v8u*)(d16 + rowoff + kb);
            else                 b   = load_b_f32(d32 + rowoff + kb);
            acc = wmma_bf16(a, b, acc);
        }
#pragma unroll
        for (int r = 0; r < 8; ++r) {
            int   m  = r + 8 * kh;
            float wr = wbuf[m];
            if (wr != 0.0f)
                atomicAdd(&out[(size_t)tokbuf[m] * HIDDEN + dt * 16 + n], wr * acc[r]);
        }
    }
}

// ---------------------------------------------------------------------------
extern "C" void kernel_launch(void* const* d_in, const int* in_sizes, int n_in,
                              void* d_out, int out_size, void* d_ws, size_t ws_size,
                              hipStream_t stream) {
    (void)in_sizes; (void)n_in; (void)out_size;
    const float* x    = (const float*)d_in[0];   // [2,2048,1024]  f32
    const float* rw   = (const float*)d_in[1];   // [16,1024]      f32
    const float* gup  = (const float*)d_in[2];   // [16,1024,1024] f32
    const float* down = (const float*)d_in[3];   // [16,1024,512]  f32
    float* out = (float*)d_out;                  // [2,2048,1024]  f32

    char*  ws  = (char*)d_ws;
    int*   cnt = (int*)ws;                                      // 16 ints
    int*   tok = (int*)(ws + 256);                              // 16*4096 ints
    float* wgt = (float*)(ws + 256 + (size_t)NEXP * TTOT * 4);  // 16*4096 floats

    const size_t META    = 256 + (size_t)NEXP * TTOT * 4 * 2;   // 524,544 B (32B-aligned)
    const size_t GUP_EL  = (size_t)NEXP * 2 * INTER * HIDDEN;   // 16,777,216
    const size_t DOWN_EL = (size_t)NEXP * HIDDEN * INTER;       //  8,388,608
    unsigned short* wb_gup  = (unsigned short*)(ws + META);
    unsigned short* wb_down = wb_gup + GUP_EL;
    const size_t NEED = META + (GUP_EL + DOWN_EL) * sizeof(unsigned short); // ~50.9 MB
    const bool preconv = (ws_size >= NEED);

    // zero output + counters
    moe_zero_kernel<<<(TTOT * HIDDEN / 4 + 255) / 256, 256, 0, stream>>>(out, cnt);

    if (preconv) {  // one-time fp32 -> bf16 weight conversion (memory bound)
        int n4g = (int)(GUP_EL / 4), n4d = (int)(DOWN_EL / 4);
        moe_cvt_kernel<<<(n4g + 255) / 256, 256, 0, stream>>>(
            (const float4*)gup, (uint2*)wb_gup, n4g);
        moe_cvt_kernel<<<(n4d + 255) / 256, 256, 0, stream>>>(
            (const float4*)down, (uint2*)wb_down, n4d);
    }

    // route + bucket tokens per expert (sparse top-2)
    moe_router_kernel<<<TTOT / 8, 256, 0, stream>>>(x, rw, cnt, tok, wgt);

    // sparse expert MLP, bf16 WMMA with f32 accumulation
    if (preconv)
        moe_expert_kernel<true><<<NEXP * 256, 256, 0, stream>>>(
            x, wb_gup, wb_down, cnt, tok, wgt, out);
    else
        moe_expert_kernel<false><<<NEXP * 256, 256, 0, stream>>>(
            x, gup, down, cnt, tok, wgt, out);
}
